// KnowledgeAttention_83313775608167
// MI455X (gfx1250) — compile-verified, matrix operand order checked by reading, not run on hardware
//
// MI455X (gfx1250, CDNA5, wave32) KnowledgeAttention.
// Pipeline: (1) W -> bf16 transposed, (2) double-buffered bf16-WMMA GEMMs for
// q/k/v projections (async global->LDS for weight tiles when available),
// (3) per-(b,lq) attention core with LDS-staged K, (4) WMMA GEMM @ Wo.
#include <hip/hip_runtime.h>

typedef __attribute__((ext_vector_type(16))) __bf16 v16bf;
typedef __attribute__((ext_vector_type(8)))  float  v8f;
typedef int v4i __attribute__((ext_vector_type(4)));

#define B_SZ  4
#define LQ    256
#define LK    64
#define QS    1024
#define H     16
#define DH    64

#define M_TILE 128
#define N_TILE 128
#define K_STEP 32
#define LDSROW 40   // 32 bf16 + 8 pad => 80B rows, 16B aligned, bank-spread

#if defined(__has_builtin)
#if __has_builtin(__builtin_amdgcn_global_load_async_to_lds_b128)
#define KA_HAVE_ASYNC 1
#endif
#endif
#ifndef KA_HAVE_ASYNC
#define KA_HAVE_ASYNC 0
#endif

// Builtin parameter types (from hipcc diagnostic): arg0 = AS1 (global) v4i*,
// arg1 = AS3 (LDS) v4i*, then two int immediates (offset, cpol).
typedef v4i __attribute__((address_space(1)))* gv4i_p;
typedef v4i __attribute__((address_space(3)))* sv4i_p;

static __device__ __forceinline__ unsigned short f2bf(float x) {
    unsigned u = __float_as_uint(x);
    unsigned r = u + 0x7FFFu + ((u >> 16) & 1u);   // round-to-nearest-even
    return (unsigned short)(r >> 16);
}
static __device__ __forceinline__ float bf2f(unsigned short h) {
    return __uint_as_float(((unsigned)h) << 16);
}

union FragU {
    uint4 u4[2];
    v16bf v;
};

// ---------------------------------------------------------------------------
// Convert W [K=1024][N=1024] f32 row-major -> Wt [N][K] bf16 row-major.
// ---------------------------------------------------------------------------
__global__ __launch_bounds__(256)
void ka_convert_transpose_w(const float* __restrict__ W,
                            unsigned short* __restrict__ Wt) {
    __shared__ unsigned short tile[32][33];
    int bx = blockIdx.x * 32;   // n-origin
    int by = blockIdx.y * 32;   // k-origin
    int tx = threadIdx.x;       // 0..31
    int ty = threadIdx.y;       // 0..7
    #pragma unroll
    for (int i = 0; i < 32; i += 8) {
        float v = W[(size_t)(by + ty + i) * QS + (bx + tx)];
        tile[ty + i][tx] = f2bf(v);
    }
    __syncthreads();
    #pragma unroll
    for (int i = 0; i < 32; i += 8) {
        Wt[(size_t)(bx + ty + i) * QS + (by + tx)] = tile[tx][ty + i];
    }
}

// ---------------------------------------------------------------------------
// C[M,1024] = A[M,1024](f32) x Bt[1024,1024]^T   (Bt stored [N][K] bf16)
// 256 threads = 8 wave32s; block tile 128x128; each wave: 64x32 via 4x2 WMMA
// accumulators. Double-buffered LDS; B-tile uses async global->LDS if present.
// ---------------------------------------------------------------------------
template <bool OUT_BF16>
__global__ __launch_bounds__(256)
void ka_gemm_bf16(const float* __restrict__ A,
                  const unsigned short* __restrict__ Bt,
                  void* __restrict__ C,
                  int M) {
    (void)M;
    __shared__ unsigned short As[2][M_TILE * LDSROW];
    __shared__ unsigned short Bs[2][N_TILE * LDSROW];

    const int Kd = QS, Nd = QS;
    const int tid  = threadIdx.x;
    const int m0   = blockIdx.x * M_TILE;
    const int n0   = blockIdx.y * N_TILE;
    const int wave = tid >> 5;
    const int lane = tid & 31;
    const int wm   = (wave >> 2) * 64;  // 0 / 64
    const int wn   = (wave & 3) * 32;   // 0 / 32 / 64 / 96
    const int lr   = lane & 15;
    const int g    = lane >> 4;         // half-wave group selects K sub-blocks

    v8f acc[4][2];
    #pragma unroll
    for (int mf = 0; mf < 4; ++mf)
        #pragma unroll
        for (int nf = 0; nf < 2; ++nf)
            #pragma unroll
            for (int i = 0; i < 8; ++i)
                acc[mf][nf][i] = 0.0f;

    float4 areg[4];
#if !KA_HAVE_ASYNC
    uint4 breg[2];
#endif

    // ---- global fetch of A tile (f32) into registers
    auto loadA_g = [&](int kk) {
        #pragma unroll
        for (int it = 0; it < 4; ++it) {
            int idx = tid + it * 256;
            int r   = idx >> 3;
            int c4  = (idx & 7) * 4;
            areg[it] = *(const float4*)(A + (size_t)(m0 + r) * Kd + kk + c4);
        }
    };
    // ---- convert + store A tile into LDS buffer
    auto storeA = [&](int buf) {
        #pragma unroll
        for (int it = 0; it < 4; ++it) {
            int idx = tid + it * 256;
            int r   = idx >> 3;
            int c4  = (idx & 7) * 4;
            unsigned p0 = (unsigned)f2bf(areg[it].x) | ((unsigned)f2bf(areg[it].y) << 16);
            unsigned p1 = (unsigned)f2bf(areg[it].z) | ((unsigned)f2bf(areg[it].w) << 16);
            *(uint2*)(&As[buf][r * LDSROW + c4]) = make_uint2(p0, p1);
        }
    };
    // ---- B tile: 128 n-rows x 32 k (bf16 already in [N][K] layout)
    auto loadB = [&](int buf, int kk) {
        #pragma unroll
        for (int it = 0; it < 2; ++it) {
            int idx = tid + it * 256;
            int r   = idx >> 2;
            int c8  = (idx & 3) * 8;
#if KA_HAVE_ASYNC
            __builtin_amdgcn_global_load_async_to_lds_b128(
                (gv4i_p)(const void*)(Bt + (size_t)(n0 + r) * Kd + kk + c8),
                (sv4i_p)(void*)(&Bs[buf][r * LDSROW + c8]),
                0, 0);
#else
            breg[it] = *(const uint4*)(Bt + (size_t)(n0 + r) * Kd + kk + c8);
#endif
        }
    };
#if !KA_HAVE_ASYNC
    auto storeB = [&](int buf) {
        #pragma unroll
        for (int it = 0; it < 2; ++it) {
            int idx = tid + it * 256;
            int r   = idx >> 2;
            int c8  = (idx & 3) * 8;
            *(uint4*)(&Bs[buf][r * LDSROW + c8]) = breg[it];
        }
    };
#endif

    // ---- preamble: stage tile 0
    loadA_g(0);
    loadB(0, 0);
    storeA(0);
#if !KA_HAVE_ASYNC
    storeB(0);
#else
    asm volatile("s_wait_asynccnt 0x0" ::: "memory");
#endif
    __syncthreads();

    const int NK = Kd / K_STEP;   // 32
    for (int k = 0; k < NK; ++k) {
        const int cur  = k & 1;
        const int nxt  = cur ^ 1;
        const bool more = (k + 1) < NK;

        // prefetch next tile while current computes
        if (more) {
            loadA_g((k + 1) * K_STEP);
            loadB(nxt, (k + 1) * K_STEP);
        }

        // --- fragments from LDS[cur]: elem j -> K = 16*(j/8)+(j%8)+8*g
        v16bf af[4], bfr[2];
        #pragma unroll
        for (int mf = 0; mf < 4; ++mf) {
            int r = wm + mf * 16 + lr;
            FragU fu;
            fu.u4[0] = *(const uint4*)(&As[cur][r * LDSROW + 8 * g]);       // K=8g..
            fu.u4[1] = *(const uint4*)(&As[cur][r * LDSROW + 16 + 8 * g]);  // K=16+8g..
            af[mf] = fu.v;
        }
        #pragma unroll
        for (int nf = 0; nf < 2; ++nf) {
            int c = wn + nf * 16 + lr;
            FragU fu;
            fu.u4[0] = *(const uint4*)(&Bs[cur][c * LDSROW + 8 * g]);
            fu.u4[1] = *(const uint4*)(&Bs[cur][c * LDSROW + 16 + 8 * g]);
            bfr[nf] = fu.v;
        }
        #pragma unroll
        for (int mf = 0; mf < 4; ++mf)
            #pragma unroll
            for (int nf = 0; nf < 2; ++nf)
                acc[mf][nf] = __builtin_amdgcn_wmma_f32_16x16x32_bf16(
                    false, af[mf], false, bfr[nf], (short)0, acc[mf][nf],
                    false, false);

        // commit prefetched tile into the other buffer
        if (more) {
            storeA(nxt);
#if !KA_HAVE_ASYNC
            storeB(nxt);
#endif
        }
#if KA_HAVE_ASYNC
        asm volatile("s_wait_asynccnt 0x0" ::: "memory");
#endif
        __syncthreads();
    }

    // --- epilogue: C/D layout: elem i -> row 8*g+i, col = lr
    #pragma unroll
    for (int mf = 0; mf < 4; ++mf) {
        #pragma unroll
        for (int nf = 0; nf < 2; ++nf) {
            #pragma unroll
            for (int i = 0; i < 8; ++i) {
                int row = m0 + wm + mf * 16 + g * 8 + i;
                int col = n0 + wn + nf * 16 + lr;
                if (OUT_BF16)
                    ((unsigned short*)C)[(size_t)row * Nd + col] = f2bf(acc[mf][nf][i]);
                else
                    ((float*)C)[(size_t)row * Nd + col] = acc[mf][nf][i];
            }
        }
    }
}

// ---------------------------------------------------------------------------
// Attention core: one block per (b,lq). 16 heads x 64 keys.
// qp [B*LQ,1024] bf16; kp/vp [B*LQ*64,1024] bf16; bias [B,1,LQ,LK] f32.
// align_out [B,H,LK,LQ] f32; ctx_out [B*LQ,1024] f32 (pre-Wo).
// K is staged through LDS in 16-key chunks so the 128MB stream is coalesced.
// ---------------------------------------------------------------------------
#define KCH   16                 // keys per staged chunk
#define KCROW (QS + 8)           // padded row: spreads banks for dot reads

__global__ __launch_bounds__(256)
void ka_attention(const unsigned short* __restrict__ qp,
                  const unsigned short* __restrict__ kp,
                  const unsigned short* __restrict__ vp,
                  const float* __restrict__ bias,
                  float* __restrict__ align_out,
                  float* __restrict__ ctx_out) {
    const int bq  = blockIdx.x;          // b*LQ + lq
    const int b   = bq >> 8;
    const int lq  = bq & 255;
    const int tid = threadIdx.x;

    __shared__ float sq[QS];
    __shared__ float sw[H * LK];
    __shared__ unsigned short skc[KCH * KCROW];   // ~33KB staged K chunk

    for (int i = tid; i < QS; i += 256)
        sq[i] = bf2f(qp[(size_t)bq * QS + i]);

    const float scale = 0.125f;  // (KS/H)^-0.5 = 64^-0.5

    for (int kb = 0; kb < LK; kb += KCH) {
        __syncthreads();
        // stage KCH x 1024 bf16 coalesced: 2048 uint4 / 256 threads = 8 each
        #pragma unroll
        for (int it = 0; it < 8; ++it) {
            int idx = tid + it * 256;
            int kk  = idx >> 7;          // 128 uint4 per key row
            int c8  = (idx & 127) * 8;
            *(uint4*)(&skc[kk * KCROW + c8]) =
                *(const uint4*)(kp + ((size_t)bq * LK + kb + kk) * QS + c8);
        }
        __syncthreads();
        // 256 threads = 16 heads x 16 keys; 64-long dot from LDS
        int h  = tid >> 4;
        int kk = tid & 15;
        float acc = 0.f;
        #pragma unroll 8
        for (int d = 0; d < DH; ++d)
            acc += sq[h * DH + d] * bf2f(skc[kk * KCROW + h * DH + d]);
        sw[h * LK + kb + kk] =
            acc * scale + bias[((size_t)b * LQ + lq) * LK + (kb + kk)];
    }
    __syncthreads();

    if (tid < H) {
        float m = -1e30f;
        for (int kk = 0; kk < LK; ++kk) m = fmaxf(m, sw[tid * LK + kk]);
        float s = 0.f;
        for (int kk = 0; kk < LK; ++kk) {
            float e = __expf(sw[tid * LK + kk] - m);
            sw[tid * LK + kk] = e;
            s += e;
        }
        float inv = 1.f / s;
        for (int kk = 0; kk < LK; ++kk) sw[tid * LK + kk] *= inv;
    }
    __syncthreads();

    // align = weights^T over (k,q): [B,H,LK,LQ]
    for (int p = tid; p < H * LK; p += 256) {
        int h  = p >> 6;
        int kk = p & 63;
        align_out[(((size_t)b * H + h) * LK + kk) * LQ + lq] = sw[p];
    }

    // context: ctx[h,d] = sum_kk w[h,kk] * v[kk, h*DH+d]; kk-outer => lanes
    // with consecutive d hit consecutive addresses (coalesced 128B bursts)
    for (int p = tid; p < QS; p += 256) {
        int h = p >> 6;
        int d = p & 63;
        const unsigned short* vcol = vp + (size_t)bq * LK * QS + h * DH + d;
        float acc = 0.f;
        #pragma unroll 8
        for (int kk = 0; kk < LK; ++kk)
            acc += sw[h * LK + kk] * bf2f(vcol[(size_t)kk * QS]);
        ctx_out[(size_t)bq * QS + p] = acc;
    }
}

// ---------------------------------------------------------------------------
extern "C" void kernel_launch(void* const* d_in, const int* in_sizes, int n_in,
                              void* d_out, int out_size, void* d_ws, size_t ws_size,
                              hipStream_t stream) {
    (void)in_sizes; (void)n_in; (void)out_size; (void)ws_size;

    const float* queries = (const float*)d_in[0];
    const float* keys    = (const float*)d_in[1];
    const float* values  = (const float*)d_in[2];
    const float* bias    = (const float*)d_in[3];
    const float* Wq      = (const float*)d_in[4];
    const float* Wk      = (const float*)d_in[5];
    const float* Wv      = (const float*)d_in[6];
    const float* Wo      = (const float*)d_in[7];

    char* ws = (char*)d_ws;
    const size_t MB = 1024ull * 1024ull;
    unsigned short* WqT = (unsigned short*)(ws + 0 * MB);
    unsigned short* WkT = (unsigned short*)(ws + 2 * MB);
    unsigned short* WvT = (unsigned short*)(ws + 4 * MB);
    unsigned short* WoT = (unsigned short*)(ws + 6 * MB);
    unsigned short* qP  = (unsigned short*)(ws + 8 * MB);    // [1024,1024] bf16
    float*          ctx = (float*)(ws + 10 * MB);            // [1024,1024] f32
    unsigned short* kP  = (unsigned short*)(ws + 16 * MB);   // [65536,1024] bf16
    unsigned short* vP  = (unsigned short*)(ws + 144 * MB);  // [65536,1024] bf16

    // 1) weight convert + transpose to bf16 [N][K]
    dim3 tb(32, 8), tg(32, 32);
    ka_convert_transpose_w<<<tg, tb, 0, stream>>>(Wq, WqT);
    ka_convert_transpose_w<<<tg, tb, 0, stream>>>(Wk, WkT);
    ka_convert_transpose_w<<<tg, tb, 0, stream>>>(Wv, WvT);
    ka_convert_transpose_w<<<tg, tb, 0, stream>>>(Wo, WoT);

    // 2) projections (WMMA bf16, f32 accum)
    const int Mq = B_SZ * LQ;              // 1024
    const int Mk = B_SZ * LQ * LK;         // 65536
    dim3 gq(Mq / M_TILE, QS / N_TILE);     // 8 x 8
    dim3 gkv(Mk / M_TILE, QS / N_TILE);    // 512 x 8
    ka_gemm_bf16<true><<<gq,  256, 0, stream>>>(queries, WqT, qP, Mq);
    ka_gemm_bf16<true><<<gkv, 256, 0, stream>>>(keys,    WkT, kP, Mk);
    ka_gemm_bf16<true><<<gkv, 256, 0, stream>>>(values,  WvT, vP, Mk);

    // 3) attention core -> align (output #2) + pre-Wo context
    float* out_ctx   = (float*)d_out;
    float* out_align = (float*)d_out + (size_t)B_SZ * LQ * QS;
    ka_attention<<<B_SZ * LQ, 256, 0, stream>>>(qP, kP, vP, bias, out_align, ctx);

    // 4) output projection -> f32 context (output #1)
    ka_gemm_bf16<false><<<gq, 256, 0, stream>>>(ctx, WoT, out_ctx, Mq);
}
